// ApsMultiheadAttention_21423296873180
// MI455X (gfx1250) — matrix operand order repaired and checked by reading, not compiled, not run on hardware
//
#include <hip/hip_runtime.h>

// ---------------------------------------------------------------------------
// MI455X (gfx1250) fused multi-head attention, f16 WMMA with f32 accumulate.
// L=1024, N=8, E=1024, H=16, D=64.  d_out = [context (L,N,E) | attn (N,L,L)].
// GEMMs use CDNA5 async global->LDS staging (ASYNCcnt) with double buffering.
// ---------------------------------------------------------------------------

typedef __attribute__((ext_vector_type(16))) _Float16 v16h;
typedef __attribute__((ext_vector_type(8)))  float    v8f;

#define L_DIM 1024
#define N_B   8
#define E_DIM 1024
#define H_DIM 16
#define D_DIM 64
#define E3    3072

// ---- CDNA5 async global->LDS (tracked by ASYNCcnt) -------------------------

__device__ __forceinline__ void async_load_b128(unsigned lds_off,
                                                const void* gaddr) {
  asm volatile("global_load_async_to_lds_b128 %0, %1, off"
               :
               : "v"(lds_off), "v"((unsigned long long)(uintptr_t)gaddr)
               : "memory");
}

__device__ __forceinline__ void wait_async0() {
  asm volatile("s_wait_asynccnt 0" ::: "memory");
}

// Flat shared-aperture address: low 32 bits are the LDS byte offset.
__device__ __forceinline__ unsigned lds_offset(const void* p) {
  return (unsigned)(uintptr_t)p;
}

// ---- WMMA fragment helpers (contiguous k = 16*(lane/16) + s, wave32) ----------

// A-matrix 16x32 f16: lanes 0-15 = rows, lanes 16-31 hold K=16..31.
__device__ __forceinline__ v16h load_frag_a(const _Float16* base, int row0,
                                            int k0, int ldk) {
  const int lane = threadIdx.x & 31;
  const int row  = row0 + (lane & 15);
  const int kb   = (lane >> 4) << 4;  // 0 or 16
  const _Float16* p = base + (size_t)row * ldk + k0;
  v16h a;
#pragma unroll
  for (int v = 0; v < 8; ++v) {
    const int k = kb + 2 * v;
    a[2 * v]     = p[k];
    a[2 * v + 1] = p[k + 1];
  }
  return a;
}

// B-matrix 32x16 f16 from [n][k] storage: lane = column, K contiguous,
// lanes 16-31 hold K=16..31.  16 halves = 32B contiguous -> b128 loads.
__device__ __forceinline__ v16h load_frag_b(const _Float16* base, int n0,
                                            int k0, int ldk) {
  const int lane = threadIdx.x & 31;
  const int n    = n0 + (lane & 15);
  const int kb   = (lane >> 4) << 4;  // 0 or 16
  return *(const v16h*)(base + (size_t)n * ldk + k0 + kb);
}

__device__ __forceinline__ v8f wmma_f16(v16h a, v16h b, v8f c) {
  return __builtin_amdgcn_wmma_f32_16x16x32_f16(false, a, false, b, (short)0,
                                                c, false, false);
}

// ---- fp32 -> f16 conversion ------------------------------------------------

__global__ __launch_bounds__(256) void f32_to_f16_kernel(
    const float* __restrict__ src, _Float16* __restrict__ dst, int n) {
  const int i = (blockIdx.x * 256 + threadIdx.x) * 4;
  if (i + 3 < n) {
    const float4 v = *(const float4*)(src + i);
    dst[i + 0] = (_Float16)v.x;
    dst[i + 1] = (_Float16)v.y;
    dst[i + 2] = (_Float16)v.z;
    dst[i + 3] = (_Float16)v.w;
  }
}

// ---- Tiled WMMA GEMM: C = A @ Bw^T + bias ---------------------------------
// A: [M][K] f16 row-major, Bw: [Nn][K] f16 row-major (weight layout).
// Block tile 128x128, BK=32, 8 waves each computing 64x32.
// Double-buffered LDS; next tile staged with async global->LDS b128 while
// WMMAs consume the current one; single barrier per k-step.

template <bool F16OUT>
__global__ __launch_bounds__(256) void wmma_gemm_nt(
    const _Float16* __restrict__ A, const _Float16* __restrict__ Bw,
    const float* __restrict__ bias, _Float16* __restrict__ Ch,
    float* __restrict__ Cf, int M, int Nn, int K) {
  __shared__ _Float16 As[2][128 * 32];
  __shared__ _Float16 Bs[2][128 * 32];

  const int tid  = threadIdx.x;
  const int lane = tid & 31;
  const int wave = tid >> 5;
  const int m0 = blockIdx.x * 128;
  const int n0 = blockIdx.y * 128;
  const int wm = (wave >> 2) * 64;  // 0 or 64
  const int wn = (wave & 3) * 32;   // 0,32,64,96
  const int lrow = tid >> 1;        // 0..127
  const int lcol = (tid & 1) * 16;  // 0 or 16 halves

  // Per-thread global source pointers (k advances by 32 halves per step).
  const _Float16* ga = A  + (size_t)(m0 + lrow) * K + lcol;
  const _Float16* gb = Bw + (size_t)(n0 + lrow) * K + lcol;
  const unsigned aoffs[2] = {lds_offset(&As[0][lrow * 32 + lcol]),
                             lds_offset(&As[1][lrow * 32 + lcol])};
  const unsigned boffs[2] = {lds_offset(&Bs[0][lrow * 32 + lcol]),
                             lds_offset(&Bs[1][lrow * 32 + lcol])};

  // Prologue: stage k=0 tile into buffer 0.
  async_load_b128(aoffs[0],      ga);
  async_load_b128(aoffs[0] + 16, (const char*)ga + 16);
  async_load_b128(boffs[0],      gb);
  async_load_b128(boffs[0] + 16, (const char*)gb + 16);
  wait_async0();
  __syncthreads();

  v8f acc[4][2] = {};
  int buf = 0;

  for (int k0 = 0; k0 < K; k0 += 32) {
    const int nbuf = buf ^ 1;
    if (k0 + 32 < K) {  // issue async stage of the next tile, no wait yet
      const char* gan = (const char*)(ga + k0 + 32);
      const char* gbn = (const char*)(gb + k0 + 32);
      async_load_b128(aoffs[nbuf],      gan);
      async_load_b128(aoffs[nbuf] + 16, gan + 16);
      async_load_b128(boffs[nbuf],      gbn);
      async_load_b128(boffs[nbuf] + 16, gbn + 16);
    }

    const _Float16* Ab = As[buf];
    const _Float16* Bb = Bs[buf];
    const v16h bf0 = load_frag_b(Bb, wn, 0, 32);
    const v16h bf1 = load_frag_b(Bb, wn + 16, 0, 32);
#pragma unroll
    for (int i = 0; i < 4; ++i) {
      const v16h af = load_frag_a(Ab, wm + i * 16, 0, 32);
      acc[i][0] = wmma_f16(af, bf0, acc[i][0]);
      acc[i][1] = wmma_f16(af, bf1, acc[i][1]);
    }

    wait_async0();      // next tile fully landed in LDS
    __syncthreads();    // all waves done reading current buffer
    buf = nbuf;
  }

  const int rbase = (lane >> 4) << 3;
  const int cn    = lane & 15;
#pragma unroll
  for (int i = 0; i < 4; ++i)
#pragma unroll
    for (int j = 0; j < 2; ++j)
#pragma unroll
      for (int r = 0; r < 8; ++r) {
        const int row = m0 + wm + i * 16 + rbase + r;
        const int col = n0 + wn + j * 16 + cn;
        const float v = acc[i][j][r] + bias[col];
        if constexpr (F16OUT)
          Ch[(size_t)row * Nn + col] = (_Float16)v;
        else
          Cf[(size_t)row * Nn + col] = v;
      }
}

// ---- Fused attention: per (n, 16-row l-tile), loop over heads -------------
// S=QK^T*0.125 (WMMA, S in registers), softmax (shfl + LDS reductions),
// attn mean accumulated in registers, P->LDS f16, O=PV (WMMA per-wave
// s-chunk + cross-wave LDS reduce), context -> f16 workspace.

__global__ __launch_bounds__(256) void attention_kernel(
    const _Float16* __restrict__ qkv,  // [8192][3072] (Q|K|V)
    _Float16* __restrict__ ch,         // [8192][1024] context (f16)
    float* __restrict__ attn_out) {    // [N][L][L]
  extern __shared__ char smem[];
  _Float16* Pl  = (_Float16*)(smem);           // 16 x 1024      (32 KB)
  _Float16* Vt  = (_Float16*)(smem + 32768);   // 8 x (64 x 32)  (32 KB)
  _Float16* Qs  = (_Float16*)(smem + 65536);   // 16 x 64        ( 2 KB)
  float*    redA = (float*)(smem + 67584);     // 8 x 16
  float*    redB = (float*)(smem + 68096);     // 8 x 16
  float*    Ored = (float*)(smem + 68608);     // 16 x 64        ( 4 KB)

  const int tid   = threadIdx.x;
  const int lane  = tid & 31;
  const int wave  = tid >> 5;
  const int l0    = blockIdx.x * 16;
  const int n     = blockIdx.y;
  const int scol0 = wave * 128;          // this wave's 128 s-columns
  const int rbase = (lane >> 4) << 3;
  const int cn    = lane & 15;

  _Float16* VtW = Vt + wave * (64 * 32);
  v8f attn_acc[8] = {};                  // head-mean of softmax weights

  for (int h = 0; h < H_DIM; ++h) {
    // ---- stage Q tile (16 x 64) into LDS ----
    {
      const int idx = tid * 4;           // 1024 halves total
      const int qr = idx >> 6, qc = idx & 63;
      const _Float16* src =
          qkv + (size_t)((l0 + qr) * N_B + n) * E3 + h * D_DIM + qc;
      *(uint2*)&Qs[idx] = *(const uint2*)src;
    }
    __syncthreads();

    // ---- S = Q K^T * (1/sqrt(D)) ----
    v8f acc[8] = {};
    const v16h aq0 = load_frag_a(Qs, 0, 0, 64);
    const v16h aq1 = load_frag_a(Qs, 0, 32, 64);
#pragma unroll
    for (int t = 0; t < 8; ++t) {
      const int s = scol0 + t * 16 + cn;
      // K row s, head h; B-fragment straight from global (32B contiguous).
      const _Float16* kbase = qkv + (size_t)(s * N_B + n) * E3 + E_DIM +
                              h * D_DIM + ((lane >> 4) << 4);
      const v16h b0 = *(const v16h*)(kbase);
      const v16h b1 = *(const v16h*)(kbase + 32);
      acc[t] = wmma_f16(aq0, b0, acc[t]);
      acc[t] = wmma_f16(aq1, b1, acc[t]);
    }
#pragma unroll
    for (int t = 0; t < 8; ++t)
#pragma unroll
      for (int r = 0; r < 8; ++r) acc[t][r] *= 0.125f;  // exact in f32

    // ---- row max (16 lanes per row-half, then across 8 waves) ----
    float m[8];
#pragma unroll
    for (int r = 0; r < 8; ++r) {
      float v = acc[0][r];
#pragma unroll
      for (int t = 1; t < 8; ++t) v = fmaxf(v, acc[t][r]);
#pragma unroll
      for (int off = 8; off >= 1; off >>= 1)
        v = fmaxf(v, __shfl_xor(v, off, 16));
      m[r] = v;
    }
    if ((lane & 15) == 0) {
#pragma unroll
      for (int r = 0; r < 8; ++r) redA[wave * 16 + rbase + r] = m[r];
    }
    __syncthreads();
    float rmax[8];
#pragma unroll
    for (int r = 0; r < 8; ++r) {
      float v = redA[rbase + r];
      for (int w = 1; w < 8; ++w) v = fmaxf(v, redA[w * 16 + rbase + r]);
      rmax[r] = v;
    }

    // ---- exp + row sum ----
#pragma unroll
    for (int r = 0; r < 8; ++r) {
      float s = 0.f;
#pragma unroll
      for (int t = 0; t < 8; ++t) {
        const float p = __expf(acc[t][r] - rmax[r]);
        acc[t][r] = p;
        s += p;
      }
#pragma unroll
      for (int off = 8; off >= 1; off >>= 1) s += __shfl_xor(s, off, 16);
      m[r] = s;
    }
    if ((lane & 15) == 0) {
#pragma unroll
      for (int r = 0; r < 8; ++r) redB[wave * 16 + rbase + r] = m[r];
    }
    __syncthreads();
#pragma unroll
    for (int r = 0; r < 8; ++r) {
      float s = 0.f;
      for (int w = 0; w < 8; ++w) s += redB[w * 16 + rbase + r];
      m[r] = 1.0f / s;
    }

    // ---- normalize, accumulate attn mean, spill P (f16) to LDS ----
#pragma unroll
    for (int t = 0; t < 8; ++t)
#pragma unroll
      for (int r = 0; r < 8; ++r) {
        const float p = acc[t][r] * m[r];
        attn_acc[t][r] += p * (1.0f / H_DIM);
        Pl[(rbase + r) * 1024 + scol0 + t * 16 + cn] = (_Float16)p;
      }
    __syncthreads();

    // ---- O = P V over this wave's s-chunk (V transposed through LDS) ----
    v8f oacc[4] = {};
    for (int ks = 0; ks < 4; ++ks) {
      const int s = scol0 + ks * 32 + lane;
      const _Float16* vrow =
          qkv + (size_t)(s * N_B + n) * E3 + 2 * E_DIM + h * D_DIM;
#pragma unroll
      for (int d = 0; d < 64; ++d) VtW[d * 32 + lane] = vrow[d];  // transpose
      const v16h ap = load_frag_a(Pl, 0, scol0 + ks * 32, 1024);
#pragma unroll
      for (int t = 0; t < 4; ++t) {
        const v16h bf = load_frag_b(VtW, t * 16, 0, 32);
        oacc[t] = wmma_f16(ap, bf, oacc[t]);
      }
    }

    // ---- cross-wave reduce O (16x64) via LDS, write context f16 ----
#pragma unroll
    for (int i = 0; i < 4; ++i) Ored[tid + i * 256] = 0.f;
    __syncthreads();
#pragma unroll
    for (int t = 0; t < 4; ++t)
#pragma unroll
      for (int r = 0; r < 8; ++r)
        atomicAdd(&Ored[(rbase + r) * 64 + t * 16 + cn], oacc[t][r]);
    __syncthreads();
#pragma unroll
    for (int i = 0; i < 4; ++i) {
      const int idx = tid + i * 256;
      const int row = idx >> 6, d = idx & 63;
      ch[(size_t)((l0 + row) * N_B + n) * E_DIM + h * D_DIM + d] =
          (_Float16)Ored[idx];
    }
    __syncthreads();
  }

  // ---- store attn mean: attn[n][l][s] ----
  float* abase = attn_out + (size_t)n * L_DIM * L_DIM + (size_t)l0 * L_DIM;
#pragma unroll
  for (int t = 0; t < 8; ++t)
#pragma unroll
    for (int r = 0; r < 8; ++r)
      abase[(size_t)(rbase + r) * L_DIM + scol0 + t * 16 + cn] =
          attn_acc[t][r];
}

// ---------------------------------------------------------------------------

extern "C" void kernel_launch(void* const* d_in, const int* in_sizes, int n_in,
                              void* d_out, int out_size, void* d_ws,
                              size_t ws_size, hipStream_t stream) {
  (void)in_sizes; (void)n_in; (void)out_size; (void)ws_size;
  const float* x     = (const float*)d_in[0];  // (L,N,E)
  const float* w_in  = (const float*)d_in[1];  // (3E,E)
  const float* b_in  = (const float*)d_in[2];  // (3E)
  const float* w_out = (const float*)d_in[3];  // (E,E)
  const float* b_out = (const float*)d_in[4];  // (E)
  float* out = (float*)d_out;

  char* ws = (char*)d_ws;
  _Float16* xh   = (_Float16*)(ws);                       // 16 MB
  _Float16* wh   = (_Float16*)(ws + 16777216);            //  6 MB
  _Float16* owh  = (_Float16*)(ws + 23068672);            //  2 MB
  _Float16* qkvh = (_Float16*)(ws + 25165824);            // 48 MB
  _Float16* chh  = (_Float16*)(ws + 75497472);            // 16 MB

  // fp32 -> f16
  f32_to_f16_kernel<<<dim3(8192), 256, 0, stream>>>(x,     xh,  8388608);
  f32_to_f16_kernel<<<dim3(3072), 256, 0, stream>>>(w_in,  wh,  3145728);
  f32_to_f16_kernel<<<dim3(1024), 256, 0, stream>>>(w_out, owh, 1048576);

  // qkv = x @ in_proj_weight^T + bias   (8192 x 3072 x 1024) -> f16
  wmma_gemm_nt<true><<<dim3(64, 24), 256, 0, stream>>>(
      xh, wh, b_in, qkvh, nullptr, 8192, 3072, 1024);

  // fused softmax(QK^T/8)V + attn head-mean
  attention_kernel<<<dim3(64, 8), 256, 72704, stream>>>(
      qkvh, chh, out + 8388608);

  // context = ctx @ out_w^T + out_b     (8192 x 1024 x 1024) -> f32 d_out
  wmma_gemm_nt<false><<<dim3(64, 8), 256, 0, stream>>>(
      chh, owh, b_out, nullptr, out, 8192, 1024, 1024);
}
